// INEncoder_35854386987246
// MI455X (gfx1250) — matrix operand order repaired and checked by reading, not compiled, not run on hardware
//
#include <hip/hip_runtime.h>

// ---------------------------------------------------------------------------
// Interaction-Network encoder for MI455X (gfx1250, wave32, WMMA f16->f32).
// Edge/node MLP GEMMs run on v_wmma_f32_16x16x32_f16 with f32 accumulation.
// ---------------------------------------------------------------------------

typedef __attribute__((ext_vector_type(16))) _Float16 v16h;
typedef __attribute__((ext_vector_type(8)))  float    v8f;

namespace {
constexpr int Bb = 128;   // batch
constexpr int Nn = 64;    // nodes
constexpr int Ll = 64;    // latent
constexpr int Cc = 2;     // channels
constexpr int Gg = 3;     // graph layers
constexpr int Ee = 4032;  // edges (N*(N-1))
constexpr int Ss = 8;     // edge-kernel segments per (c,b)
constexpr float LN_EPS = 1e-5f;
constexpr int PIN = 200;  // padded LDS row stride (halves) for F<=192 inputs / W0t
constexpr int PW  = 72;   // padded LDS row stride (halves) for 64-wide tiles
}

// ---- WMMA fragment loaders (contiguous k order, k = 16*half + j, for A and B, wave32) ---------

// A: 16x32 f16 tile from LDS (row-major, 'stride' halves per row), K-chunk kc.
__device__ __forceinline__ v16h a_frag(const _Float16* base, int stride, int kc) {
  const int lane = threadIdx.x & 31;
  const int m = lane & 15, hh = lane >> 4;
  const _Float16* p = base + m * stride + kc * 32 + hh * 16;
  v16h a;
#pragma unroll
  for (int j = 0; j < 16; ++j) a[j] = p[j];         // K = hh*16 + j
  return a;
}

// B: 32x16 f16 tile; weights stored transposed in LDS: bt[n][k].
__device__ __forceinline__ v16h b_frag(const _Float16* bt, int stride, int nt, int kc) {
  const int lane = threadIdx.x & 31;
  const int n = lane & 15, g = lane >> 4;
  const _Float16* p = bt + (nt * 16 + n) * stride + kc * 32 + g * 16;
  v16h b;
#pragma unroll
  for (int j = 0; j < 16; ++j) b[j] = p[j];         // K = g*16 + j
  return b;
}

// One wave computes a 16xF @ Fx64 GEMM -> 4 accumulator tiles of 16x16 f32.
__device__ __forceinline__ void gemm_16xF_64(const _Float16* aBase, int aStride, int F,
                                             const _Float16* bt, int bStride,
                                             const float* bias, bool relu, v8f acc[4]) {
  const int n0 = threadIdx.x & 15;
  const int kcN = F >> 5;
#pragma unroll
  for (int nt = 0; nt < 4; ++nt) {
    v8f cacc;
    const float bv = bias[nt * 16 + n0];
#pragma unroll
    for (int r = 0; r < 8; ++r) cacc[r] = bv;
    for (int kc = 0; kc < kcN; ++kc) {
      v16h a = a_frag(aBase, aStride, kc);
      v16h b = b_frag(bt, bStride, nt, kc);
      cacc = __builtin_amdgcn_wmma_f32_16x16x32_f16(false, a, false, b,
                                                    (short)0, cacc, false, false);
    }
    if (relu) {
#pragma unroll
      for (int r = 0; r < 8; ++r) cacc[r] = fmaxf(cacc[r], 0.0f);
    }
    acc[nt] = cacc;
  }
}

// C/D layout: reg r -> M = r + 8*(lane>=16), N = lane&15.
__device__ __forceinline__ void store_tile_f16(_Float16* dst, int stride, const v8f acc[4]) {
  const int lane = threadIdx.x & 31;
  const int n0 = lane & 15, mh = (lane >> 4) * 8;
#pragma unroll
  for (int nt = 0; nt < 4; ++nt)
#pragma unroll
    for (int r = 0; r < 8; ++r)
      dst[(mh + r) * stride + nt * 16 + n0] = (_Float16)acc[nt][r];
}

// ---------------------------------------------------------------------------
// Fused edge MLP: gather(x[dst],x[src],LN(edges)) -> 3 GEMMs -> edges out,
// LDS-accumulated scatter-add partials, per-batch LN statistics.
// Grid: (Ss segments, B, C); block: 128 threads (4 waves).
// ---------------------------------------------------------------------------
__global__ __launch_bounds__(128)
void edge_kernel(const float* __restrict__ x, int x_chan,
                 const int* __restrict__ src, const int* __restrict__ dst,
                 const float* __restrict__ W0, int W0cs, int F0,
                 const float* __restrict__ b0,
                 const float* __restrict__ W1, const float* __restrict__ b1,
                 const float* __restrict__ W2, const float* __restrict__ b2,
                 float* edges, int has_edges_in, int store_edges,
                 const float* __restrict__ e_mu, const float* __restrict__ e_rsig,
                 const float* __restrict__ lnw, const float* __restrict__ lnb,
                 float* __restrict__ aggPart, float* __restrict__ estats) {
  __shared__ _Float16 sW0t[64 * PIN];
  __shared__ _Float16 sW1t[64 * PW];
  __shared__ _Float16 sW2t[64 * PW];
  __shared__ float sB0[64], sB1[64], sB2[64];
  __shared__ _Float16 sIn[4 * 16 * PIN];
  __shared__ _Float16 sH0[4 * 16 * PW];
  __shared__ float sAgg[Nn * Ll];

  const int c = blockIdx.z, b = blockIdx.y;
  const int tid = threadIdx.x, wave = tid >> 5, lane = tid & 31;

  // Stage weights (transposed, f16) and biases; zero the agg tile.
  const float* W0c = W0 + (size_t)c * W0cs;
  const float* W1c = W1 + (size_t)c * (Ll * Ll);
  const float* W2c = W2 + (size_t)c * (Ll * Ll);
  for (int idx = tid; idx < F0 * 64; idx += 128) {
    int k = idx >> 6, n = idx & 63;
    sW0t[n * PIN + k] = (_Float16)W0c[k * 64 + n];
  }
  for (int idx = tid; idx < 64 * 64; idx += 128) {
    int k = idx >> 6, n = idx & 63;
    sW1t[n * PW + k] = (_Float16)W1c[k * 64 + n];
    sW2t[n * PW + k] = (_Float16)W2c[k * 64 + n];
  }
  if (tid < 64) {
    sB0[tid] = b0[(size_t)c * Ll + tid];
    sB1[tid] = b1[(size_t)c * Ll + tid];
    sB2[tid] = b2[(size_t)c * Ll + tid];
  }
  for (int idx = tid; idx < Nn * Ll; idx += 128) sAgg[idx] = 0.0f;
  __syncthreads();

  const float* xb = x + (x_chan ? (size_t)(c * Bb + b) : (size_t)b) * Nn * Ll;
  float* eb = edges + ((size_t)(c * Bb + b)) * Ee * Ll;
  const float* lnwc = lnw + (size_t)c * Ee * Ll;
  const float* lnbc = lnb + (size_t)c * Ee * Ll;
  float emu = 0.0f, ersig = 1.0f;
  if (has_edges_in) { emu = e_mu[b]; ersig = e_rsig[b]; }

  _Float16* in = &sIn[wave * 16 * PIN];
  _Float16* h0 = &sH0[wave * 16 * PW];
  const int cpr = F0 >> 2;  // float4 chunks per row
  float ssum = 0.0f, ssq = 0.0f;

  for (int tt = blockIdx.x; tt < Ee / 64; tt += Ss) {
    const int ebase = tt * 64 + wave * 16;

    // ---- stage 16 gathered rows as f16 (LN applied to incoming edges) ----
    for (int q = lane; q < 16 * cpr; q += 32) {
      const int j = q / cpr, col = (q % cpr) * 4;
      const int e = ebase + j;
      float4 v;
      if (col < 64) {
        const int t = dst[e];
        v = *(const float4*)(xb + (size_t)t * 64 + col);
      } else if (col < 128) {
        const int s2 = src[e];
        v = *(const float4*)(xb + (size_t)s2 * 64 + (col - 64));
      } else {
        const size_t o = (size_t)e * 64 + (col - 128);
        const float4 ev = *(const float4*)(eb + o);
        const float4 wv = *(const float4*)(lnwc + o);
        const float4 bv = *(const float4*)(lnbc + o);
        v.x = (ev.x - emu) * ersig * wv.x + bv.x;
        v.y = (ev.y - emu) * ersig * wv.y + bv.y;
        v.z = (ev.z - emu) * ersig * wv.z + bv.z;
        v.w = (ev.w - emu) * ersig * wv.w + bv.w;
      }
      _Float16* d = in + j * PIN + col;
      d[0] = (_Float16)v.x; d[1] = (_Float16)v.y;
      d[2] = (_Float16)v.z; d[3] = (_Float16)v.w;
    }
    __syncthreads();

    // ---- 3-layer MLP via WMMA ----
    v8f acc[4];
    gemm_16xF_64(in, PIN, F0, sW0t, PIN, sB0, true, acc);
    store_tile_f16(h0, PW, acc);
    __syncthreads();
    gemm_16xF_64(h0, PW, 64, sW1t, PW, sB1, true, acc);
    store_tile_f16(in, PW, acc);  // reuse sIn region as h1
    __syncthreads();
    gemm_16xF_64(in, PW, 64, sW2t, PW, sB2, false, acc);

    // ---- epilogue: edges out, LDS scatter-add, LN stats ----
    const int n0 = lane & 15, mh = (lane >> 4) * 8;
    int de[8];
#pragma unroll
    for (int r = 0; r < 8; ++r) de[r] = dst[ebase + mh + r];
#pragma unroll
    for (int nt = 0; nt < 4; ++nt) {
#pragma unroll
      for (int r = 0; r < 8; ++r) {
        const float v = acc[nt][r];
        const int e = ebase + mh + r;
        if (store_edges) eb[(size_t)e * 64 + nt * 16 + n0] = v;
        atomicAdd(&sAgg[de[r] * 64 + nt * 16 + n0], v);
        ssum += v; ssq += v * v;
      }
    }
    __syncthreads();
  }

  // Per-segment aggregation partial (no global atomics).
  float* ap = aggPart + (((size_t)blockIdx.x * Cc + c) * Bb + b) * (Nn * Ll);
  for (int idx = tid; idx < Nn * Ll; idx += 128) ap[idx] = sAgg[idx];

  float t1 = ssum, t2 = ssq;
  for (int m = 16; m > 0; m >>= 1) {
    t1 += __shfl_xor(t1, m, 32);
    t2 += __shfl_xor(t2, m, 32);
  }
  if (lane == 0) {
    atomicAdd(&estats[2 * b], t1);
    atomicAdd(&estats[2 * b + 1], t2);
  }
}

// ---------------------------------------------------------------------------
// Fused node MLP: concat(x, sum_s aggPart) -> 3 GEMMs -> x_out + LN stats.
// Grid: (B, C); block: 128 threads (4 waves, one 16-row tile each).
// ---------------------------------------------------------------------------
__global__ __launch_bounds__(128)
void node_kernel(const float* __restrict__ x, int x_chan,
                 const float* __restrict__ aggPart,
                 const float* __restrict__ W0, const float* __restrict__ b0,
                 const float* __restrict__ W1, const float* __restrict__ b1,
                 const float* __restrict__ W2, const float* __restrict__ b2,
                 float* __restrict__ xout, float* __restrict__ nstats) {
  constexpr int PINN = 136;  // 128 + 8 pad
  __shared__ _Float16 sW0t[64 * PINN];
  __shared__ _Float16 sW1t[64 * PW];
  __shared__ _Float16 sW2t[64 * PW];
  __shared__ float sB0[64], sB1[64], sB2[64];
  __shared__ _Float16 sIn[4 * 16 * PINN];
  __shared__ _Float16 sH0[4 * 16 * PW];

  const int b = blockIdx.x, c = blockIdx.y;
  const int tid = threadIdx.x, wave = tid >> 5, lane = tid & 31;

  const float* W0c = W0 + (size_t)c * (2 * Ll * Ll);
  const float* W1c = W1 + (size_t)c * (Ll * Ll);
  const float* W2c = W2 + (size_t)c * (Ll * Ll);
  for (int idx = tid; idx < 128 * 64; idx += 128) {
    int k = idx >> 6, n = idx & 63;
    sW0t[n * PINN + k] = (_Float16)W0c[k * 64 + n];
  }
  for (int idx = tid; idx < 64 * 64; idx += 128) {
    int k = idx >> 6, n = idx & 63;
    sW1t[n * PW + k] = (_Float16)W1c[k * 64 + n];
    sW2t[n * PW + k] = (_Float16)W2c[k * 64 + n];
  }
  if (tid < 64) {
    sB0[tid] = b0[(size_t)c * Ll + tid];
    sB1[tid] = b1[(size_t)c * Ll + tid];
    sB2[tid] = b2[(size_t)c * Ll + tid];
  }
  __syncthreads();

  const float* xb = x + (x_chan ? (size_t)(c * Bb + b) : (size_t)b) * Nn * Ll;
  const float* ab = aggPart + ((size_t)c * Bb + b) * (Nn * Ll);
  const size_t sstride = (size_t)Cc * Bb * Nn * Ll;

  // ---- stage 16 rows: [x | sum_s aggPart] as f16 ----
  _Float16* in = &sIn[wave * 16 * PINN];
  for (int q = lane; q < 16 * 32; q += 32) {
    const int j = q >> 5, col = (q & 31) * 4;
    const int n = wave * 16 + j;
    float4 v;
    if (col < 64) {
      v = *(const float4*)(xb + (size_t)n * 64 + col);
    } else {
      v.x = v.y = v.z = v.w = 0.0f;
#pragma unroll
      for (int s = 0; s < Ss; ++s) {
        const float4 t = *(const float4*)(ab + s * sstride + (size_t)n * 64 + (col - 64));
        v.x += t.x; v.y += t.y; v.z += t.z; v.w += t.w;
      }
    }
    _Float16* d = in + j * PINN + col;
    d[0] = (_Float16)v.x; d[1] = (_Float16)v.y;
    d[2] = (_Float16)v.z; d[3] = (_Float16)v.w;
  }
  __syncthreads();

  v8f acc[4];
  _Float16* h0 = &sH0[wave * 16 * PW];
  gemm_16xF_64(in, PINN, 128, sW0t, PINN, sB0, true, acc);
  store_tile_f16(h0, PW, acc);
  __syncthreads();
  gemm_16xF_64(h0, PW, 64, sW1t, PW, sB1, true, acc);
  store_tile_f16(in, PW, acc);
  __syncthreads();
  gemm_16xF_64(in, PW, 64, sW2t, PW, sB2, false, acc);

  // ---- epilogue: write x_out, accumulate LN stats ----
  const int n0 = lane & 15, mh = (lane >> 4) * 8;
  float ssum = 0.0f, ssq = 0.0f;
  float* xo = xout + ((size_t)(c * Bb + b)) * Nn * Ll;
#pragma unroll
  for (int nt = 0; nt < 4; ++nt) {
#pragma unroll
    for (int r = 0; r < 8; ++r) {
      const float v = acc[nt][r];
      const int n = wave * 16 + mh + r;
      xo[(size_t)n * 64 + nt * 16 + n0] = v;
      ssum += v; ssq += v * v;
    }
  }
  for (int m = 16; m > 0; m >>= 1) {
    ssum += __shfl_xor(ssum, m, 32);
    ssq += __shfl_xor(ssq, m, 32);
  }
  if (lane == 0) {
    atomicAdd(&nstats[2 * b], ssum);
    atomicAdd(&nstats[2 * b + 1], ssq);
  }
}

// ---------------------------------------------------------------------------
// Small helper kernels.
// ---------------------------------------------------------------------------
__global__ void zero_f32(float* __restrict__ p, int n) {
  const int i = blockIdx.x * 256 + threadIdx.x;
  if (i < n) p[i] = 0.0f;
}

__global__ void finalize_stats(const float* __restrict__ stats,
                               float* __restrict__ mu, float* __restrict__ rsig,
                               float cnt) {
  const int b = threadIdx.x;
  if (b < Bb) {
    const float m = stats[2 * b] / cnt;
    float v = stats[2 * b + 1] / cnt - m * m;
    v = fmaxf(v, 0.0f);
    mu[b] = m;
    rsig[b] = rsqrtf(v + LN_EPS);
  }
}

__global__ void ln_apply_nodes(float* __restrict__ x,
                               const float* __restrict__ mu, const float* __restrict__ rsig,
                               const float* __restrict__ w, const float* __restrict__ bv) {
  const size_t i = (size_t)blockIdx.x * 256 + threadIdx.x;  // over C*B*N*L
  const int l = (int)(i & 63);
  size_t t = i >> 6;
  const int n = (int)(t & 63); t >>= 6;
  const int b = (int)(t & 127);
  const int c = (int)(t >> 7);
  const size_t wi = (((size_t)c * Nn + n) * Ll + l);
  x[i] = (x[i] - mu[b]) * rsig[b] * w[wi] + bv[wi];
}

__global__ void sum_channels(const float* __restrict__ x, float* __restrict__ out) {
  const size_t i = (size_t)blockIdx.x * 256 + threadIdx.x;  // over B*N*L
  out[i] = x[i] + x[i + (size_t)Bb * Nn * Ll];
}

// ---------------------------------------------------------------------------
extern "C" void kernel_launch(void* const* d_in, const int* in_sizes, int n_in,
                              void* d_out, int out_size, void* d_ws, size_t ws_size,
                              hipStream_t stream) {
  (void)in_sizes; (void)n_in; (void)out_size; (void)ws_size;
  const float* nodes     = (const float*)d_in[0];
  const int*   src       = (const int*)d_in[1];
  const int*   dst       = (const int*)d_in[2];
  const float* eW0_first = (const float*)d_in[3];
  const float* eb0_first = (const float*)d_in[4];
  const float* eW0_rest  = (const float*)d_in[5];
  const float* eb0_rest  = (const float*)d_in[6];
  const float* eW1 = (const float*)d_in[7];
  const float* eb1 = (const float*)d_in[8];
  const float* eW2 = (const float*)d_in[9];
  const float* eb2 = (const float*)d_in[10];
  const float* nW0 = (const float*)d_in[11];
  const float* nb0 = (const float*)d_in[12];
  const float* nW1 = (const float*)d_in[13];
  const float* nb1 = (const float*)d_in[14];
  const float* nW2 = (const float*)d_in[15];
  const float* nb2 = (const float*)d_in[16];
  const float* ln_nw = (const float*)d_in[17];
  const float* ln_nb = (const float*)d_in[18];
  const float* ln_ew = (const float*)d_in[19];
  const float* ln_eb = (const float*)d_in[20];

  float* ws = (float*)d_ws;
  size_t off = 0;
  float* edges   = ws + off; off += (size_t)Cc * Bb * Ee * Ll;       // 66.1 M f32
  float* xw      = ws + off; off += (size_t)Cc * Bb * Nn * Ll;       // 1.05 M f32
  float* aggPart = ws + off; off += (size_t)Ss * Cc * Bb * Nn * Ll;  // 8.4 M f32
  float* estats  = ws + off; off += 2 * Bb;
  float* nstats  = ws + off; off += 2 * Bb;
  float* e_mu    = ws + off; off += Bb;
  float* e_rsig  = ws + off; off += Bb;
  float* n_mu    = ws + off; off += Bb;
  float* n_rsig  = ws + off; off += Bb;

  for (int i = 0; i < Gg; ++i) {
    zero_f32<<<2, 256, 0, stream>>>(estats, 4 * Bb);  // estats+nstats contiguous

    const int F0 = (i == 0) ? 2 * Ll : 3 * Ll;
    const float* W0  = (i == 0) ? eW0_first : eW0_rest + (size_t)(i - 1) * Cc * 3 * Ll * Ll;
    const float* b0p = (i == 0) ? eb0_first : eb0_rest + (size_t)(i - 1) * Cc * Ll;

    edge_kernel<<<dim3(Ss, Bb, Cc), 128, 0, stream>>>(
        (i == 0) ? nodes : xw, (i > 0) ? 1 : 0, src, dst,
        W0, F0 * Ll, F0, b0p,
        eW1 + (size_t)i * Cc * Ll * Ll, eb1 + (size_t)i * Cc * Ll,
        eW2 + (size_t)i * Cc * Ll * Ll, eb2 + (size_t)i * Cc * Ll,
        edges, (i > 0) ? 1 : 0, (i < Gg - 1) ? 1 : 0,
        e_mu, e_rsig,
        (i > 0) ? ln_ew + (size_t)(i - 1) * Cc * Ee * Ll : ln_ew,
        (i > 0) ? ln_eb + (size_t)(i - 1) * Cc * Ee * Ll : ln_eb,
        aggPart, estats);

    finalize_stats<<<1, 128, 0, stream>>>(estats, e_mu, e_rsig,
                                          (float)((size_t)Cc * Ee * Ll));

    node_kernel<<<dim3(Bb, Cc), 128, 0, stream>>>(
        (i == 0) ? nodes : xw, (i > 0) ? 1 : 0, aggPart,
        nW0 + (size_t)i * Cc * 2 * Ll * Ll, nb0 + (size_t)i * Cc * Ll,
        nW1 + (size_t)i * Cc * Ll * Ll, nb1 + (size_t)i * Cc * Ll,
        nW2 + (size_t)i * Cc * Ll * Ll, nb2 + (size_t)i * Cc * Ll,
        xw, nstats);

    finalize_stats<<<1, 128, 0, stream>>>(nstats, n_mu, n_rsig,
                                          (float)(Cc * Nn * Ll));

    ln_apply_nodes<<<(Cc * Bb * Nn * Ll) / 256, 256, 0, stream>>>(
        xw, n_mu, n_rsig,
        ln_nw + (size_t)i * Cc * Nn * Ll, ln_nb + (size_t)i * Cc * Nn * Ll);
  }

  sum_channels<<<(Bb * Nn * Ll) / 256, 256, 0, stream>>>(xw, (float*)d_out);
}
